// TransposableGene_25185688223999
// MI455X (gfx1250) — matrix operand order repaired and checked by reading, not compiled, not run on hardware
//
#include <hip/hip_runtime.h>

typedef __attribute__((ext_vector_type(16))) _Float16 v16h;
typedef __attribute__((ext_vector_type(8)))  float    v8f;

#define H_DIM 128
#define LN_EPS 1e-5f

__device__ __forceinline__ float wave_reduce_sum(float v) {
#pragma unroll
  for (int off = 16; off >= 1; off >>= 1)
    v += __shfl_xor(v, off, 32);
  return v;
}

// ---------------------------------------------------------------------------
// out[M x 128] = X[M x K] * W[K x 128]   (f16 WMMA, f32 accumulate)
// One workgroup = 8 waves -> 16 rows x 128 cols; each wave does a 16x16 tile.
// GUARD path is branch-free: clamped row pointer + 0/1 mask multiply.
// ---------------------------------------------------------------------------
template <int K, bool GUARD>
__device__ __forceinline__ void gemm_tile_body(const float* __restrict__ X,
                                               const float* __restrict__ W,
                                               float* __restrict__ out, int M,
                                               int tileM, int tileN) {
  const int lane = threadIdx.x & 31;
  const int m    = lane & 15;
  const int grp  = lane >> 4;  // 0 or 1
  const int row  = tileM + m;
  const int rowL = GUARD ? (row < M ? row : M - 1) : row;
  const float mask = GUARD ? (row < M ? 1.0f : 0.0f) : 1.0f;
  const int col  = tileN + m;  // B-matrix column for this lane

  v8f c = {};
#pragma unroll
  for (int k0 = 0; k0 < K; k0 += 32) {
    // A fragment: 16x32 f16.  lane<16: K = 0..7,16..23 ; lane>=16: +8
    v16h a;
    {
      const float* xr = X + (size_t)rowL * K + k0 + grp * 8;
#pragma unroll
      for (int j = 0; j < 8; ++j) {
        const int off = (j < 4) ? (2 * j) : (16 + 2 * (j - 4));
        float2 p = *(const float2*)(xr + off);
        if (GUARD) { p.x *= mask; p.y *= mask; }
        a[2 * j]     = (_Float16)p.x;
        a[2 * j + 1] = (_Float16)p.y;
      }
    }
    // B fragment: 32x16 f16. lane<16: K = 0..7,16..23 ; lane>=16: +8
    v16h b;
    {
      const float* wr = W + (size_t)(k0 + grp * 8) * H_DIM + col;
#pragma unroll
      for (int j = 0; j < 8; ++j) {
        const int off = (j < 4) ? (2 * j) : (16 + 2 * (j - 4));
        b[2 * j]     = (_Float16)wr[off * H_DIM];
        b[2 * j + 1] = (_Float16)wr[(off + 1) * H_DIM];
      }
    }
    c = __builtin_amdgcn_wmma_f32_16x16x32_f16(false, a, false, b, (short)0, c,
                                               false, false);
  }
  // C/D layout: VGPR r -> (lane<16: M=r, lane>=16: M=8+r), N = lane&15
#pragma unroll
  for (int r = 0; r < 8; ++r) {
    const int orow = tileM + r + grp * 8;
    if (!GUARD || orow < M)
      out[(size_t)orow * H_DIM + tileN + m] = c[r];
  }
}

template <int K>
__global__ void gemm_wmma_kernel(const float* __restrict__ X,
                                 const float* __restrict__ W,
                                 float* __restrict__ out, int M) {
  const int tileM = blockIdx.x * 16;
  const int tileN = (threadIdx.x >> 5) * 16;  // 8 waves cover 128 columns
  if (tileM + 16 <= M)  // uniform branch: full tiles take the clean path
    gemm_tile_body<K, false>(X, W, out, M, tileM, tileN);
  else
    gemm_tile_body<K, true>(X, W, out, M, tileM, tileN);
}

// Runtime-K fallback (only used if feature dim != 64)
__global__ void gemm_wmma_generic_kernel(const float* __restrict__ X,
                                         const float* __restrict__ W,
                                         float* __restrict__ out, int M,
                                         int K) {
  const int lane  = threadIdx.x & 31;
  const int m     = lane & 15;
  const int grp   = lane >> 4;
  const int tileM = blockIdx.x * 16;
  const int tileN = (threadIdx.x >> 5) * 16;
  const int row   = tileM + m;
  const int rowL  = row < M ? row : M - 1;
  const float mask = row < M ? 1.0f : 0.0f;
  const int col   = tileN + m;

  v8f c = {};
  for (int k0 = 0; k0 < K; k0 += 32) {
    v16h a;
    const float* xr = X + (size_t)rowL * K + k0 + grp * 8;
#pragma unroll
    for (int j = 0; j < 8; ++j) {
      const int off = (j < 4) ? (2 * j) : (16 + 2 * (j - 4));
      float2 p = *(const float2*)(xr + off);
      p.x *= mask; p.y *= mask;
      a[2 * j]     = (_Float16)p.x;
      a[2 * j + 1] = (_Float16)p.y;
    }
    v16h b;
    const float* wr = W + (size_t)(k0 + grp * 8) * H_DIM + col;
#pragma unroll
    for (int j = 0; j < 8; ++j) {
      const int off = (j < 4) ? (2 * j) : (16 + 2 * (j - 4));
      b[2 * j]     = (_Float16)wr[off * H_DIM];
      b[2 * j + 1] = (_Float16)wr[(off + 1) * H_DIM];
    }
    c = __builtin_amdgcn_wmma_f32_16x16x32_f16(false, a, false, b, (short)0, c,
                                               false, false);
  }
#pragma unroll
  for (int r = 0; r < 8; ++r) {
    const int orow = tileM + r + grp * 8;
    if (orow < M) out[(size_t)orow * H_DIM + tileN + m] = c[r];
  }
}

// ---------------------------------------------------------------------------
// Degree / init kernels
// ---------------------------------------------------------------------------
__global__ void init_kernel(float* __restrict__ deg, float* __restrict__ counts,
                            float* __restrict__ out, int N, int G) {
  const int i = blockIdx.x * blockDim.x + threadIdx.x;
  if (i < N) deg[i] = 1.0f;  // self-loop contribution
  if (i < G * H_DIM) out[i] = 0.0f;
  if (i < G) counts[i] = 0.0f;
}

__global__ void deg_accum_kernel(const int* __restrict__ dst,
                                 float* __restrict__ deg, int E) {
  const int i = blockIdx.x * blockDim.x + threadIdx.x;
  if (i < E) atomicAdd(&deg[dst[i]], 1.0f);
}

__global__ void dinv_kernel(float* __restrict__ deg, int N) {
  const int i = blockIdx.x * blockDim.x + threadIdx.x;
  if (i < N) deg[i] = rsqrtf(fmaxf(deg[i], 1e-12f));
}

// agg[i,:] = dinv[i] * hw[i,:]   (self-loop term, dinv[d] factored out later)
__global__ void selfloop_init_kernel(const float* __restrict__ hw,
                                     const float* __restrict__ dinv,
                                     float* __restrict__ agg, int N) {
  const int t     = blockIdx.x * blockDim.x + threadIdx.x;  // one float4
  const int total = N * (H_DIM / 4);
  if (t >= total) return;
  const float  di = dinv[t >> 5];  // 32 quads per row
  const float4 v  = ((const float4*)hw)[t];
  float4 o;
  o.x = v.x * di; o.y = v.y * di; o.z = v.z * di; o.w = v.w * di;
  ((float4*)agg)[t] = o;
}

// One wave per edge: agg[d,:] += dinv[s] * hw[s,:]  (float4 per lane)
__global__ void edge_scatter_kernel(const int* __restrict__ src,
                                    const int* __restrict__ dst,
                                    const float* __restrict__ dinv,
                                    const float* __restrict__ hw,
                                    float* __restrict__ agg, int E) {
  const int lane = threadIdx.x & 31;
  const int wid  = (int)((blockIdx.x * blockDim.x + threadIdx.x) >> 5);
  const int nw   = (int)((gridDim.x * blockDim.x) >> 5);
  for (int e = wid; e < E; e += nw) {
    const int    s  = src[e];
    const int    d  = dst[e];
    const float  ns = dinv[s];
    const float4 v  = *(const float4*)(hw + (size_t)s * H_DIM + lane * 4);
    float*       ap = agg + (size_t)d * H_DIM + lane * 4;
    atomicAdd(ap + 0, v.x * ns);
    atomicAdd(ap + 1, v.y * ns);
    atomicAdd(ap + 2, v.z * ns);
    atomicAdd(ap + 3, v.w * ns);
  }
}

// Fused: h = dinv[i]*agg[i] + bias ; LayerNorm(gamma,beta) ; optional ReLU.
// One wave per row (4 floats/lane), 8 rows per 256-thread block.
template <bool RELU>
__global__ void ln_kernel(const float* __restrict__ agg,
                          const float* __restrict__ dinv,
                          const float* __restrict__ bias,
                          const float* __restrict__ gamma,
                          const float* __restrict__ beta,
                          float* __restrict__ out, int N) {
  const int lane = threadIdx.x & 31;
  const int wave = threadIdx.x >> 5;
  const int row  = blockIdx.x * 8 + wave;
  if (row >= N) return;

  const float  di = dinv[row];
  float4       v  = *(const float4*)(agg + (size_t)row * H_DIM + lane * 4);
  const float4 b4 = *(const float4*)(bias + lane * 4);
  v.x = v.x * di + b4.x; v.y = v.y * di + b4.y;
  v.z = v.z * di + b4.z; v.w = v.w * di + b4.w;

  const float mean = wave_reduce_sum(v.x + v.y + v.z + v.w) * (1.0f / H_DIM);
  const float dx = v.x - mean, dy = v.y - mean, dz = v.z - mean,
              dw = v.w - mean;
  const float var =
      wave_reduce_sum(dx * dx + dy * dy + dz * dz + dw * dw) * (1.0f / H_DIM);
  const float rstd = rsqrtf(var + LN_EPS);

  const float4 g4  = *(const float4*)(gamma + lane * 4);
  const float4 be4 = *(const float4*)(beta + lane * 4);
  float4 o;
  o.x = dx * rstd * g4.x + be4.x;
  o.y = dy * rstd * g4.y + be4.y;
  o.z = dz * rstd * g4.z + be4.z;
  o.w = dw * rstd * g4.w + be4.w;
  if (RELU) {
    o.x = fmaxf(o.x, 0.0f); o.y = fmaxf(o.y, 0.0f);
    o.z = fmaxf(o.z, 0.0f); o.w = fmaxf(o.w, 0.0f);
  }
  *(float4*)(out + (size_t)row * H_DIM + lane * 4) = o;
}

// Segment-sum pool exploiting sorted batch: each block walks a contiguous node
// chunk; thread t accumulates feature t in a register and flushes an atomic
// only when the graph id changes.
__global__ void pool_kernel(const float* __restrict__ h,
                            const int* __restrict__ batch,
                            float* __restrict__ out,
                            float* __restrict__ counts, int N) {
  const int t     = threadIdx.x;  // feature index, blockDim.x == 128
  const int chunk = (N + gridDim.x - 1) / gridDim.x;
  const int n0    = blockIdx.x * chunk;
  int       n1    = n0 + chunk;
  if (n1 > N) n1 = N;
  if (n0 >= n1) return;

  int   cur = batch[n0];
  float acc = 0.0f, cnt = 0.0f;
  for (int i = n0; i < n1; ++i) {
    const int g = batch[i];
    if (g != cur) {
      atomicAdd(&out[cur * H_DIM + t], acc);
      if (t == 0) atomicAdd(&counts[cur], cnt);
      acc = 0.0f; cnt = 0.0f; cur = g;
    }
    acc += h[(size_t)i * H_DIM + t];
    cnt += 1.0f;
  }
  atomicAdd(&out[cur * H_DIM + t], acc);
  if (t == 0) atomicAdd(&counts[cur], cnt);
}

__global__ void divide_kernel(float* __restrict__ out,
                              const float* __restrict__ counts, int G) {
  const int i = blockIdx.x * blockDim.x + threadIdx.x;
  if (i < G * H_DIM) out[i] /= fmaxf(counts[i / H_DIM], 1.0f);
}

static inline int imax(int a, int b) { return a > b ? a : b; }

extern "C" void kernel_launch(void* const* d_in, const int* in_sizes, int n_in,
                              void* d_out, int out_size, void* d_ws,
                              size_t ws_size, hipStream_t stream) {
  (void)n_in; (void)ws_size;
  const float* x   = (const float*)d_in[0];
  const int*   ei  = (const int*)d_in[1];
  const int*   bat = (const int*)d_in[2];
  const float* W1  = (const float*)d_in[3];
  const float* b1  = (const float*)d_in[4];
  const float* g1  = (const float*)d_in[5];
  const float* be1 = (const float*)d_in[6];
  const float* W2  = (const float*)d_in[7];
  const float* b2  = (const float*)d_in[8];
  const float* g2  = (const float*)d_in[9];
  const float* be2 = (const float*)d_in[10];

  const int N = in_sizes[2];       // batch has one entry per node
  const int E = in_sizes[1] / 2;   // edge_index is [2, E]
  const int F = in_sizes[0] / N;   // 64
  const int G = out_size / H_DIM;  // 128

  const int* src = ei;
  const int* dst = ei + E;

  float* A      = (float*)d_ws;           // N*H  (hw / h buffers)
  float* B      = A + (size_t)N * H_DIM;  // N*H
  float* deg    = B + (size_t)N * H_DIM;  // N    (becomes dinv)
  float* counts = deg + N;                // G
  float* out    = (float*)d_out;

  // --- degree / init ------------------------------------------------------
  const int initN = imax(N, G * H_DIM);
  init_kernel<<<(initN + 255) / 256, 256, 0, stream>>>(deg, counts, out, N, G);
  deg_accum_kernel<<<(E + 255) / 256, 256, 0, stream>>>(dst, deg, E);
  dinv_kernel<<<(N + 255) / 256, 256, 0, stream>>>(deg, N);

  const int quadBlocks = (N * (H_DIM / 4) + 255) / 256;
  const int gemmBlocks = (N + 15) / 16;

  // --- layer 1: GCNConv(x; W1,b1) -> LN(g1,beta1) -> ReLU -----------------
  if (F == 64)
    gemm_wmma_kernel<64><<<gemmBlocks, 256, 0, stream>>>(x, W1, A, N);
  else
    gemm_wmma_generic_kernel<<<gemmBlocks, 256, 0, stream>>>(x, W1, A, N, F);
  selfloop_init_kernel<<<quadBlocks, 256, 0, stream>>>(A, deg, B, N);
  edge_scatter_kernel<<<4096, 256, 0, stream>>>(src, dst, deg, A, B, E);
  ln_kernel<true><<<(N + 7) / 8, 256, 0, stream>>>(B, deg, b1, g1, be1, A, N);

  // --- layer 2: GCNConv(h; W2,b2) -> LN(g2,beta2) -------------------------
  gemm_wmma_kernel<128><<<gemmBlocks, 256, 0, stream>>>(A, W2, B, N);
  selfloop_init_kernel<<<quadBlocks, 256, 0, stream>>>(B, deg, A, N);
  edge_scatter_kernel<<<4096, 256, 0, stream>>>(src, dst, deg, B, A, E);
  ln_kernel<false><<<(N + 7) / 8, 256, 0, stream>>>(A, deg, b2, g2, be2, B, N);

  // --- mean pool per graph ------------------------------------------------
  pool_kernel<<<512, H_DIM, 0, stream>>>(B, bat, out, counts, N);
  divide_kernel<<<(G * H_DIM + 255) / 256, 256, 0, stream>>>(out, counts, G);
}
